// TpsGridGen_73632919323234
// MI455X (gfx1250) — matrix operand order repaired and checked by reading, not compiled, not run on hardware
//
#include <hip/hip_runtime.h>
#include <hip/hip_bf16.h>

// TPS grid gen as WMMA GEMM: D[49152 x 128] = basis[49152 x 32(f16)] * W[32 x 128(f16)]
//   basis[p,k] = [ U_0(p)..U_24(p), 1, gx, gy, 0,0,0,0 ]
//   W[k, 2b+c] = sum_m Li[k,m] * (theta[b, c*25+m] + (c? py[m] : px[m]))   (k<28, else 0)
//   out[b, p, c] = D[p, 2b+c]   with out flat = b*98304 + p*2 + c

typedef __attribute__((ext_vector_type(16))) _Float16 v16h;
typedef __attribute__((ext_vector_type(8)))  float    v8f;

#define NPIX   49152      // 192*256
#define NCTRL  25
#define KPAD   32
#define NCOLS  128        // 64 batches * {x,y}
#define OUT_B  98304      // 192*256*2 floats per batch

// ---------------------------------------------------------------------------
// Kernel 1: basis matrix, written directly as per-lane WMMA A operands, contiguous k order.
// A tile = 16(M) x 32(K) f16, 512 halves per tile.
// lane l = M + 16*h ;  v16h element j -> K = 16*h + j
// => pixel (tile,m) owns two contiguous 16-half chunks at lanes m and m+16.
// ---------------------------------------------------------------------------
__global__ void __launch_bounds__(256)
tps_basis(const float* __restrict__ grid, const float* __restrict__ px,
          const float* __restrict__ py, _Float16* __restrict__ Abuf) {
    int p = blockIdx.x * 256 + threadIdx.x;          // 192 blocks * 256 = 49152 exact
    float gx = grid[2 * p + 0];
    float gy = grid[2 * p + 1];

    float basis[32];
#pragma unroll
    for (int n = 0; n < NCTRL; ++n) {
        float dx = gx - px[n];
        float dy = gy - py[n];
        float d2 = dx * dx + dy * dy;
        basis[n] = (d2 == 0.0f) ? 0.0f : d2 * __logf(d2);
    }
    basis[25] = 1.0f; basis[26] = gx; basis[27] = gy;
    basis[28] = basis[29] = basis[30] = basis[31] = 0.0f;

    int tile = p >> 4;
    int m    = p & 15;
    v16h c0, c1;
#pragma unroll
    for (int j = 0; j < 16; ++j) {
        c0[j]     = (_Float16)basis[j];          // h=0: K = 0..15
        c1[j]     = (_Float16)basis[16 + j];     // h=1: K = 16..31
    }
    *(v16h*)(Abuf + tile * 512 + m * 16)        = c0;
    *(v16h*)(Abuf + tile * 512 + (m + 16) * 16) = c1;
}

// ---------------------------------------------------------------------------
// Kernel 2: W matrix (28x128, zero-padded to 32) as per-lane WMMA B operands, contiguous k order.
// B tile = 32(K) x 16(N): lane l -> col = ct*16 + (l&15), K base = (l>>4)*16,
// v16h element j -> K = Kbase + j.  Bbuf[ct*512 + l*16 + j].
// ---------------------------------------------------------------------------
__global__ void __launch_bounds__(128)
tps_w(const float* __restrict__ theta, const float* __restrict__ Li,
      const float* __restrict__ px, const float* __restrict__ py,
      _Float16* __restrict__ Bbuf) {
    int k   = blockIdx.x;        // 0..31
    int col = threadIdx.x;       // 0..127
    int b   = col >> 1;
    int c   = col & 1;

    float acc = 0.0f;
    if (k < 28) {
        const float* t = theta + b * 50 + c * NCTRL;
        const float* pc = c ? py : px;
        const float* li = Li + k * 28;
#pragma unroll
        for (int m = 0; m < NCTRL; ++m)
            acc += li[m] * (t[m] + pc[m]);
    }
    int ct = col >> 4;
    int l  = (col & 15) + ((k >> 4) << 4);
    int j  = k & 15;
    Bbuf[ct * 512 + l * 16 + j] = (_Float16)acc;
}

// ---------------------------------------------------------------------------
// Kernel 3: GEMM + transposed coalesced store.
// 8 waves/block, each wave owns one M-tile (16 pixels): 8 x v_wmma_f32_16x16x32_f16,
// D staged in LDS (16 x 128 f32 per wave), then 128B-contiguous float2 stores per batch.
// ---------------------------------------------------------------------------
__global__ void __launch_bounds__(256)
tps_gemm(const _Float16* __restrict__ Abuf, const _Float16* __restrict__ Bbuf,
         float* __restrict__ out) {
    __shared__ float lds[8 * 16 * NCOLS];            // 64 KB

    int wave = threadIdx.x >> 5;
    int lane = threadIdx.x & 31;
    int mt   = blockIdx.x * 8 + wave;                // 384 blocks * 8 = 3072 exact

    __builtin_prefetch(Bbuf, 0, 0);                  // warm shared 8KB B matrix

    v16h a = *(const v16h*)(Abuf + mt * 512 + lane * 16);

    float* tile = lds + wave * (16 * NCOLS);
    int lrow0 = (lane >> 4) << 3;                    // D: VGPR v -> M = v + 8*(lane>>4)
    int lcol  = lane & 15;                           // D: lane   -> N

#pragma unroll
    for (int ct = 0; ct < 8; ++ct) {
        v16h bm = *(const v16h*)(Bbuf + ct * 512 + lane * 16);
        v8f d = {};
        d = __builtin_amdgcn_wmma_f32_16x16x32_f16(
                /*neg_a=*/false, a, /*neg_b=*/false, bm,
                /*c_mod=*/(short)0, d, /*reuse_a=*/false, /*reuse_b=*/false);
#pragma unroll
        for (int v = 0; v < 8; ++v)
            tile[(lrow0 + v) * NCOLS + ct * 16 + lcol] = d[v];
    }

    // Store: for each batch b, rows p0..p0+15, cols (2b,2b+1) are 8B contiguous;
    // 16 lanes x float2 = 128B contiguous per batch per wave-tile.
    int p0 = mt * 16;
    int i  = lane & 15;
    int bh = lane >> 4;
#pragma unroll
    for (int it = 0; it < 32; ++it) {
        int b = it * 2 + bh;
        float2 val = *(const float2*)(tile + i * NCOLS + b * 2);
        *(float2*)(out + (size_t)b * OUT_B + (size_t)(p0 + i) * 2) = val;
    }
}

// ---------------------------------------------------------------------------
extern "C" void kernel_launch(void* const* d_in, const int* in_sizes, int n_in,
                              void* d_out, int out_size, void* d_ws, size_t ws_size,
                              hipStream_t stream) {
    (void)in_sizes; (void)n_in; (void)out_size; (void)ws_size;

    const float* theta = (const float*)d_in[0];   // (64, 50)
    const float* grid  = (const float*)d_in[1];   // (1, 192, 256, 2)
    const float* Li    = (const float*)d_in[2];   // (28, 28)
    const float* px    = (const float*)d_in[3];   // (25,)
    const float* py    = (const float*)d_in[4];   // (25,)
    float*       out   = (float*)d_out;           // (64, 192, 256, 2)

    _Float16* Abuf = (_Float16*)d_ws;             // 49152*32 halves = 3 MB
    _Float16* Bbuf = Abuf + (size_t)NPIX * KPAD;  // 4096 halves = 8 KB

    tps_w    <<<dim3(32),  dim3(128), 0, stream>>>(theta, Li, px, py, Bbuf);
    tps_basis<<<dim3(192), dim3(256), 0, stream>>>(grid, px, py, Abuf);
    tps_gemm <<<dim3(384), dim3(256), 0, stream>>>(Abuf, Bbuf, out);
}
